// CABlock_65317862637876
// MI455X (gfx1250) — matrix operand order repaired and checked by reading, not compiled, not run on hardware
//
#include <hip/hip_runtime.h>
#include <hip/hip_bf16.h>

// CDNA5 / gfx1250: wave32, WMMA 16x16x32 f16 -> f32 accumulate,
// async global->LDS staging for the attention K/V windows.
typedef __attribute__((ext_vector_type(16))) _Float16 v16h;
typedef __attribute__((ext_vector_type(8)))  _Float16 v8h;
typedef __attribute__((ext_vector_type(4)))  _Float16 v4h;
typedef __attribute__((ext_vector_type(8)))  float    v8f;
typedef __attribute__((ext_vector_type(4)))  float    v4f;

#define BB   2
#define NN   4096
#define EE   256
#define HH   8
#define HD   32
#define TAUW 32
#define SCALE 0.17677669529663687f   // 1/sqrt(32)

// ---- ISA 7.12.2 fragment element maps (wave32, 16-bit A/B, f32 C/D) ----
// A (16x32 MxK): lane L holds row M = L%16; halves 0..7 -> K = (L<16?0:8)+h,
//                halves 8..15 -> K = (L<16?16:24)+(h-8).   (two contiguous 8-half runs)
// B (32x16 KxN): lane L holds col N = L%16; same K map as A. This code loads A and B as K = (L<16?0:16)+h.
// C/D (16x16 f32): lane L holds col N = L%16; VGPR r -> row M = r + (L<16?0:8).
__device__ __forceinline__ int b_k_of(int lane, int hh) {
    return ((lane < 16) ? 0 : 16) + hh;
}
__device__ __forceinline__ v16h load_a_frag(const _Float16* row, int alo) {
    v8h lo = *(const v8h*)(row + alo);        // K = alo .. alo+7
    v8h hi = *(const v8h*)(row + alo + 8);    // K = alo+8 .. alo+15
    return __builtin_shufflevector(lo, hi, 0,1,2,3,4,5,6,7,8,9,10,11,12,13,14,15);
}

// =====================================================================
// Kernel 0: elementwise f32 -> f16 (inputs and W converted exactly once)
// =====================================================================
__global__ void cvt_f16_kernel(const float* __restrict__ src,
                               _Float16* __restrict__ dst, int n4) {
    int i = blockIdx.x * blockDim.x + threadIdx.x;
    if (i < n4) {
        v4f f = *(const v4f*)(src + 4 * (size_t)i);
        v4h o;
        o[0] = (_Float16)f[0]; o[1] = (_Float16)f[1];
        o[2] = (_Float16)f[2]; o[3] = (_Float16)f[3];
        *(v4h*)(dst + 4 * (size_t)i) = o;
    }
}

// =====================================================================
// Kernel 1: projections  Qp/Kp/Vp = f16( x @ W^T + b ).
// One wave -> 16(M) x 64(Nout) tile: one A fragment feeds 4 accumulators,
// K fully unrolled (8 chunks) -> 32 static v_wmma per wave.
// =====================================================================
__global__ void ca_proj_kernel(const _Float16* __restrict__ qh,
                               const _Float16* __restrict__ kh,
                               const _Float16* __restrict__ vh,
                               const _Float16* __restrict__ Wh,
                               const float* __restrict__ bias,
                               _Float16* __restrict__ qp,
                               _Float16* __restrict__ kp,
                               _Float16* __restrict__ vp) {
    const int lane  = threadIdx.x & 31;
    const int m0    = blockIdx.x * 16;     // row tile over B*N
    const int nb    = blockIdx.y * 64;     // 4 output-feature tiles
    const int which = blockIdx.z;          // 0=q, 1=k, 2=v

    const _Float16* x = (which == 0) ? qh : (which == 1) ? kh : vh;
    _Float16*       o = (which == 0) ? qp : (which == 1) ? kp : vp;

    const int MN  = lane & 15;
    const int alo = (lane < 16) ? 0 : 16;
    const int blo = (lane < 16) ? 0 : 16;

    v8f acc[4] = {{}, {}, {}, {}};
#pragma unroll
    for (int kc = 0; kc < EE / 32; ++kc) {
        const int k0 = kc * 32;
        v16h a = load_a_frag(x + (size_t)(m0 + MN) * EE + k0, alo);
#pragma unroll
        for (int i = 0; i < 4; ++i) {
            // B[k][n] = W[n][k]: lane's column is W row nb+16i+MN, 16 contiguous K
            v16h bf = *(const v16h*)(Wh + (size_t)(nb + 16 * i + MN) * EE + k0 + blo);
            acc[i] = __builtin_amdgcn_wmma_f32_16x16x32_f16(false, a, false, bf,
                                                            (short)0, acc[i], false, false);
        }
    }
    const int mb = (lane < 16) ? 0 : 8;
#pragma unroll
    for (int i = 0; i < 4; ++i) {
        const int   col = nb + 16 * i + MN;
        const float bv  = bias[col];
#pragma unroll
        for (int r = 0; r < 8; ++r)
            o[(size_t)(m0 + mb + r) * EE + col] = (_Float16)(acc[i][r] + bv);
    }
}

// =====================================================================
// Kernel 2: sliding-window attention for one (batch, head, 16-row tile).
// K window rows (48x32) and V window rows (64x32, zero-padded) are staged
// into LDS with global_load_async_to_lds_b128 (ASYNCcnt), out-of-range
// rows filled with the projected-padding vector (= bias) or zeros.
//   scores:  Q(16x32) x K^T(32x48)           -> 3 WMMAs
//   softmax: per-row over the 32-wide band via LDS
//   output:  Wband(16x64, zero-padded) x V(64x32) -> 4 WMMAs
// =====================================================================
__global__ void ca_attn_kernel(const _Float16* __restrict__ qp,
                               const _Float16* __restrict__ kp,
                               const _Float16* __restrict__ vp,
                               const float* __restrict__ bias,
                               float* __restrict__ out) {
    __shared__ float    sS[16 * 48];    // raw scores, row-major
    __shared__ _Float16 sW[16 * 64];    // banded softmax weights (zero-padded)
    __shared__ _Float16 sK[48 * HD];    // key window rows j=0..47
    __shared__ _Float16 sV[64 * HD];    // value window rows j=0..63 (48+ zero)

    const int lane = threadIdx.x & 31;
    const int n0   = blockIdx.x * 16;
    const int h    = blockIdx.y;
    const int b    = blockIdx.z;
    const size_t base  = (size_t)b * NN * EE;
    const int    MN    = lane & 15;
    const int    alo   = (lane < 16) ? 0 : 16;
    const int    blo   = (lane < 16) ? 0 : 16;
    const int    mbase = (lane < 16) ? 0 : 8;

    // ---- stage K window: rows j=0..47, 4 x 16B segments per row ----
    for (int c = lane; c < 48 * 4; c += 32) {
        const int j = c >> 2, s = c & 3;
        const int pos = n0 - (TAUW - 1) + j;
        const unsigned ldst = (unsigned)(uintptr_t)&sK[j * HD + s * 8];
        if (j <= 46 && pos >= 0) {
            const unsigned long long g =
                (unsigned long long)(uintptr_t)(kp + base + (size_t)pos * EE + h * HD + s * 8);
            asm volatile("global_load_async_to_lds_b128 %0, %1, off"
                         :: "v"(ldst), "v"(g) : "memory");
        } else {
            v8h f;
#pragma unroll
            for (int e = 0; e < 8; ++e)
                f[e] = (j <= 46) ? (_Float16)bias[h * HD + s * 8 + e]   // proj(0)=b
                                 : (_Float16)0.0f;
            *(v8h*)&sK[j * HD + s * 8] = f;
        }
    }
    // ---- stage V window: rows j=0..63 (rows >46 are never weighted) ----
    for (int c = lane; c < 64 * 4; c += 32) {
        const int j = c >> 2, s = c & 3;
        const int pos = n0 - (TAUW - 1) + j;
        const unsigned ldst = (unsigned)(uintptr_t)&sV[j * HD + s * 8];
        if (j <= 46 && pos >= 0) {
            const unsigned long long g =
                (unsigned long long)(uintptr_t)(vp + base + (size_t)pos * EE + h * HD + s * 8);
            asm volatile("global_load_async_to_lds_b128 %0, %1, off"
                         :: "v"(ldst), "v"(g) : "memory");
        } else {
            v8h f;
#pragma unroll
            for (int e = 0; e < 8; ++e)
                f[e] = (j <= 46 && pos < 0) ? (_Float16)bias[h * HD + s * 8 + e]
                                            : (_Float16)0.0f;
            *(v8h*)&sV[j * HD + s * 8] = f;
        }
    }

    // A fragment: projected queries (direct from global, f16)
    v16h aq = load_a_frag(qp + base + (size_t)(n0 + MN) * EE + h * HD, alo);

    asm volatile("s_wait_asynccnt 0" ::: "memory");
    __syncthreads();

    // ---- scores over the 48-wide key span (B from LDS) ----
    for (int jt = 0; jt < 3; ++jt) {
        // lane's column j fixed, K=d contiguous 16 halves of that K row
        v16h bk = *(const v16h*)&sK[(jt * 16 + MN) * HD + blo];
        v8f s = {};
        s = __builtin_amdgcn_wmma_f32_16x16x32_f16(false, aq, false, bk,
                                                   (short)0, s, false, false);
#pragma unroll
        for (int r = 0; r < 8; ++r)
            sS[(mbase + r) * 48 + jt * 16 + MN] = s[r] * SCALE;
    }
    __syncthreads();

    for (int i = lane; i < 16 * 64; i += 32) sW[i] = (_Float16)0.0f;
    __syncthreads();

    // ---- softmax: lanes 0..15, one row each (band j = row .. row+31) ----
    if (lane < 16) {
        const int row = lane;
        float mx = -1e30f;
        for (int t = 0; t < TAUW; ++t)
            mx = fmaxf(mx, sS[row * 48 + row + t]);
        float sum = 0.0f;
        for (int t = 0; t < TAUW; ++t)
            sum += __expf(sS[row * 48 + row + t] - mx);
        const float inv = 1.0f / sum;
        for (int t = 0; t < TAUW; ++t)
            sW[row * 64 + row + t] =
                (_Float16)(__expf(sS[row * 48 + row + t] - mx) * inv);
    }
    __syncthreads();

    // ---- output: banded weights (K=64, padded) x V rows from LDS ----
    v8f acc0 = {}, acc1 = {};
#pragma unroll
    for (int kc = 0; kc < 2; ++kc) {
        v16h aw;
#pragma unroll
        for (int hh = 0; hh < 16; ++hh) {
            const int K = alo + hh;
            aw[hh] = sW[MN * 64 + kc * 32 + K];
        }
#pragma unroll
        for (int dt = 0; dt < 2; ++dt) {
            v16h bv;
#pragma unroll
            for (int hh = 0; hh < 16; ++hh) {
                const int j = kc * 32 + b_k_of(lane, hh);   // K = window row
                bv[hh] = sV[j * HD + dt * 16 + MN];
            }
            v8f& acc = (dt == 0) ? acc0 : acc1;
            acc = __builtin_amdgcn_wmma_f32_16x16x32_f16(false, aw, false, bv,
                                                         (short)0, acc, false, false);
        }
    }
#pragma unroll
    for (int r = 0; r < 8; ++r) {
        const size_t row = base + (size_t)(n0 + mbase + r) * EE + h * HD;
        out[row + MN]      = acc0[r];
        out[row + 16 + MN] = acc1[r];
    }
}

extern "C" void kernel_launch(void* const* d_in, const int* in_sizes, int n_in,
                              void* d_out, int out_size, void* d_ws, size_t ws_size,
                              hipStream_t stream) {
    const float* q    = (const float*)d_in[0];
    const float* k    = (const float*)d_in[1];
    const float* v    = (const float*)d_in[2];
    const float* W    = (const float*)d_in[3];
    const float* bias = (const float*)d_in[4];
    float* out = (float*)d_out;

    const size_t elems = (size_t)BB * NN * EE;       // 2,097,152
    _Float16* qh = (_Float16*)d_ws;                  // raw f16 copies
    _Float16* kh = qh + elems;
    _Float16* vh = kh + elems;
    _Float16* Wh = vh + elems;                       // 65536
    _Float16* qp = Wh + (size_t)EE * EE;             // projected f16
    _Float16* kp = qp + elems;
    _Float16* vp = kp + elems;                       // total ~24.1 MB of ws

    // 0) one-time f32 -> f16 conversions
    const int TB = 256;
    cvt_f16_kernel<<<dim3((int)(elems / 4 + TB - 1) / TB), TB, 0, stream>>>(q, qh, (int)(elems / 4));
    cvt_f16_kernel<<<dim3((int)(elems / 4 + TB - 1) / TB), TB, 0, stream>>>(k, kh, (int)(elems / 4));
    cvt_f16_kernel<<<dim3((int)(elems / 4 + TB - 1) / TB), TB, 0, stream>>>(v, vh, (int)(elems / 4));
    cvt_f16_kernel<<<dim3((EE * EE / 4 + TB - 1) / TB), TB, 0, stream>>>(W, Wh, EE * EE / 4);

    // 1) projections: (512, 4, 3) tiles, 1 wave each, 32 WMMAs per wave
    dim3 g1(BB * NN / 16, EE / 64, 3);
    ca_proj_kernel<<<g1, 32, 0, stream>>>(qh, kh, vh, Wh, bias, qp, kp, vp);

    // 2) windowed attention: (256, 8, 2), 1 wave each
    dim3 g2(NN / 16, HH, BB);
    ca_attn_kernel<<<g2, 32, 0, stream>>>(qp, kp, vp, bias, out);
}
